// MS_Resblock_4_63986422776476
// MI455X (gfx1250) — matrix operand order repaired and checked by reading, not compiled, not run on hardware
//
#include <hip/hip_runtime.h>

typedef _Float16 h16;
typedef __attribute__((ext_vector_type(16))) _Float16 v16h;
typedef __attribute__((ext_vector_type(8)))  _Float16 v8h;
typedef __attribute__((ext_vector_type(8)))  float    v8f;
typedef __attribute__((ext_vector_type(4)))  float    v4f;
typedef __attribute__((ext_vector_type(4)))  int      v4i;

#define T_  4
#define B_  4
#define C_  64
#define H_  128
#define W_  128
#define HW_ 16384            // H_*W_
#define TB_ 16               // T_*B_
#define SP_ 262144           // TB_*HW_  (spatial elements per channel)
#define K_  576              // C_*9
#define NEL 16777216         // SP_*C_  (elements per full tensor)

// CDNA5 async global->LDS staging (ASYNCcnt path), with safe fallback.
#if defined(__AMDGCN__) && \
    __has_builtin(__builtin_amdgcn_global_load_async_to_lds_b128) && \
    __has_builtin(__builtin_amdgcn_s_wait_asynccnt)
#define USE_ASYNC_LDS 1
typedef __attribute__((address_space(1))) v4i gv4i;  // global b128 chunk
typedef __attribute__((address_space(3))) v4i lv4i;  // LDS b128 chunk
#else
#define USE_ASYNC_LDS 0
#endif

// ---------------------------------------------------------------------------
// Weight repack: fp32 [8][O][I][3][3] -> fp16 [8][O][K], K = (ky*3+kx)*64 + ci
// ---------------------------------------------------------------------------
__global__ __launch_bounds__(256) void k_prep_w(const float* __restrict__ w,
                                                h16* __restrict__ gw) {
  int i = blockIdx.x * 256 + threadIdx.x;
  if (i >= 8 * 64 * K_) return;
  int p64 = i % K_;
  int o   = (i / K_) % 64;
  int j   = i / (K_ * 64);
  int p   = p64 >> 6;        // kernel position 0..8
  int ci  = p64 & 63;
  gw[i] = (h16)w[((j * 64 + o) * 64 + ci) * 9 + p];
}

// ---------------------------------------------------------------------------
// LIF on original input x (NCHW fp32) -> spikes (NHWC fp16)
// ---------------------------------------------------------------------------
__global__ __launch_bounds__(256) void k_lif_x(const float* __restrict__ x,
                                               h16* __restrict__ spk) {
  int i  = blockIdx.x * 256 + threadIdx.x;   // over B*HW*C = 2^22
  int c  = i & 63;
  int hw = (i >> 6) & (HW_ - 1);
  int b  = i >> 20;
  float v = 0.f;
#pragma unroll
  for (int t = 0; t < T_; ++t) {
    int tb = t * B_ + b;
    float xv = x[(tb * 64 + c) * HW_ + hw];
    v += (xv - v) * 0.5f;
    bool s = (v >= 1.f);
    spk[(tb * HW_ + hw) * 64 + c] = s ? (h16)1.0f : (h16)0.0f;
    if (s) v = 0.f;
  }
}

// ---------------------------------------------------------------------------
// LIF on NHWC fp32 'out' -> spikes (NHWC fp16) + v_seq (NHWC fp32, post-reset)
// ---------------------------------------------------------------------------
__global__ __launch_bounds__(256) void k_lif_mid(const float* __restrict__ src,
                                                 h16* __restrict__ spk,
                                                 float* __restrict__ vseq) {
  int i  = blockIdx.x * 256 + threadIdx.x;
  int c  = i & 63;
  int hw = (i >> 6) & (HW_ - 1);
  int b  = i >> 20;
  float v = 0.f;
#pragma unroll
  for (int t = 0; t < T_; ++t) {
    int tb  = t * B_ + b;
    int idx = (tb * HW_ + hw) * 64 + c;
    float xv = src[idx];
    v += (xv - v) * 0.5f;
    bool s = (v >= 1.f);
    spk[idx] = s ? (h16)1.0f : (h16)0.0f;
    if (s) v = 0.f;
    vseq[idx] = v;
  }
}

// ---------------------------------------------------------------------------
// Fused TEBN-apply + LIF: h = A[t,c]*y + B[t,c]; LIF(h) -> spikes NHWC fp16
// ---------------------------------------------------------------------------
__global__ __launch_bounds__(256) void k_bn_lif(const float* __restrict__ y,
                                                const float* __restrict__ cA,
                                                const float* __restrict__ cB,
                                                h16* __restrict__ spk) {
  int i  = blockIdx.x * 256 + threadIdx.x;
  int c  = i & 63;
  int hw = (i >> 6) & (HW_ - 1);
  int b  = i >> 20;
  float v = 0.f;
#pragma unroll
  for (int t = 0; t < T_; ++t) {
    int tb  = t * B_ + b;
    int idx = (tb * HW_ + hw) * 64 + c;
    float hv = cA[t * 64 + c] * y[idx] + cB[t * 64 + c];
    v += (hv - v) * 0.5f;
    bool s = (v >= 1.f);
    spk[idx] = s ? (h16)1.0f : (h16)0.0f;
    if (s) v = 0.f;
  }
}

// ---------------------------------------------------------------------------
// Fused TEBN-apply + residual add. resid_nchw: residual layout; out_nchw: out layout
// ---------------------------------------------------------------------------
__global__ __launch_bounds__(256) void k_bn_add(const float* __restrict__ y,
                                                const float* __restrict__ cA,
                                                const float* __restrict__ cB,
                                                const float* __restrict__ resid,
                                                float* __restrict__ out,
                                                int resid_nchw, int out_nchw) {
  int i  = blockIdx.x * 256 + threadIdx.x;
  int c  = i & 63;
  int hw = (i >> 6) & (HW_ - 1);
  int b  = i >> 20;
#pragma unroll
  for (int t = 0; t < T_; ++t) {
    int tb   = t * B_ + b;
    int inh  = (tb * HW_ + hw) * 64 + c;   // NHWC index
    int inc  = (tb * 64 + c) * HW_ + hw;   // NCHW index
    float rv = resid_nchw ? resid[inc] : resid[inh];
    float ov = cA[t * 64 + c] * y[inh] + cB[t * 64 + c] + rv;
    if (out_nchw) out[inc] = ov; else out[inh] = ov;
  }
}

// ---------------------------------------------------------------------------
// BN stats pass 1: per-channel partial sum / sumsq (deterministic 2-pass)
// ---------------------------------------------------------------------------
__global__ __launch_bounds__(256) void k_stats_part(const float* __restrict__ y,
                                                    float* __restrict__ part) {
  __shared__ float ss[256];
  __shared__ float sq[256];
  int tid = threadIdx.x, c = tid & 63, g = tid >> 6;
  float s = 0.f, q = 0.f;
  for (int sp = blockIdx.x * 4 + g; sp < SP_; sp += 1024) {
    float v = y[sp * 64 + c];
    s += v; q += v * v;
  }
  ss[tid] = s; sq[tid] = q;
  __syncthreads();
  if (tid < 64) {
    s = ss[tid] + ss[tid + 64] + ss[tid + 128] + ss[tid + 192];
    q = sq[tid] + sq[tid + 64] + sq[tid + 128] + sq[tid + 192];
    part[blockIdx.x * 128 + tid]      = s;
    part[blockIdx.x * 128 + 64 + tid] = q;
  }
}

// ---------------------------------------------------------------------------
// BN stats pass 2: fold mean/invstd/gamma/beta/per-t scale into affine (A,B)
// ---------------------------------------------------------------------------
__global__ void k_stats_final(const float* __restrict__ part,
                              const float* __restrict__ gamma,
                              const float* __restrict__ beta,
                              const float* __restrict__ scale,  // 4 per-t values
                              float* __restrict__ cA, float* __restrict__ cB) {
  int c = threadIdx.x;  // 64 threads
  float S = 0.f, Q = 0.f;
  for (int i = 0; i < 256; ++i) {
    S += part[i * 128 + c];
    Q += part[i * 128 + 64 + c];
  }
  const float inv_n = 1.f / (float)SP_;
  float mean = S * inv_n;
  float var  = Q * inv_n - mean * mean;
  float inv  = rsqrtf(var + 1e-5f);
  float gc   = gamma[c] * inv;
#pragma unroll
  for (int t = 0; t < T_; ++t) {
    float k = scale[t];
    cA[t * 64 + c] = gc * k;
    cB[t * 64 + c] = (beta[c] - mean * gc) * k;
  }
}

// ---------------------------------------------------------------------------
// 3x3 conv via implicit GEMM with V_WMMA_F32_16X16X32_F16.
// One block: (tb, h, w0..w0+63) x 64 output channels.
// LDS: zero-padded input patch 3 rows x 66 cols x 64 ch (fp16, 25 KB),
// staged via GLOBAL_LOAD_ASYNC_TO_LDS_B128 when the toolchain exposes it.
// 8 waves; wave -> M-tile = 16*(wave&3), N-tiles at 32*(wave>>2) + {0,16}.
// K = 576 consumed in 18 steps of 32 (kpos-major, channel-minor).
// ---------------------------------------------------------------------------
__global__ __launch_bounds__(256) void k_conv3x3_wmma(
    const h16* __restrict__ spk, const h16* __restrict__ gw,
    float* __restrict__ y) {
  __shared__ h16 lB[3 * 66 * 64];

  const int bid  = blockIdx.x;
  const int w0   = (bid & 1) * 64;
  const int h    = (bid >> 1) & 127;
  const int tb   = bid >> 8;
  const int tid  = threadIdx.x;

  // Stage input patch with zero padding (v8h = 16B chunks)
  for (int idx = tid; idx < 3 * 66 * 8; idx += 256) {
    int c8  = idx & 7;
    int col = (idx >> 3) % 66;
    int ry  = idx / (8 * 66);
    int yy  = h - 1 + ry;
    int xx  = w0 - 1 + col;
    h16* ldst = lB + (ry * 66 + col) * 64 + c8 * 8;
    bool inb = ((unsigned)yy < (unsigned)H_) && ((unsigned)xx < (unsigned)W_);
#if USE_ASYNC_LDS
    if (inb) {
      const h16* gsrc = spk + (((tb * H_ + yy) * W_ + xx) * 64 + c8 * 8);
      __builtin_amdgcn_global_load_async_to_lds_b128(
          (gv4i*)gsrc, (lv4i*)ldst, 0, 0);
    } else {
      *reinterpret_cast<v8h*>(ldst) = (v8h){};
    }
#else
    v8h val = {};
    if (inb)
      val = *reinterpret_cast<const v8h*>(
          spk + (((tb * H_ + yy) * W_ + xx) * 64 + c8 * 8));
    *reinterpret_cast<v8h*>(ldst) = val;
#endif
  }
#if USE_ASYNC_LDS
  __builtin_amdgcn_s_wait_asynccnt(0);
#endif
  __syncthreads();

  const int wave  = tid >> 5;
  const int lane  = tid & 31;
  const int lm    = lane & 15;   // M (for A) / N (for B) index within tile
  const int lh    = lane >> 4;   // K-half selector
  const int mBase = (wave & 3) << 4;
  const int nb0   = (wave >> 2) << 5;  // 0 or 32

  v8f acc0 = {};
  v8f acc1 = {};
  const h16* wrow = gw + (mBase + lm) * K_;   // weight row for this lane's M

  for (int p = 0; p < 9; ++p) {
    const int ky = p / 3, kx = p % 3;
    const h16* brow0 = &lB[(ky * 66 + (nb0 + lm + kx)) * 64];
    const h16* brow1 = &lB[(ky * 66 + (nb0 + 16 + lm + kx)) * 64];
#pragma unroll
    for (int half = 0; half < 2; ++half) {
      const int ci0 = half * 32;
      // A fragment: lane holds K = 16*lh .. 16*lh+15 (contig channels)
      v8h alo = *reinterpret_cast<const v8h*>(wrow + p * 64 + ci0 + 16 * lh);
      v8h ahi = *reinterpret_cast<const v8h*>(wrow + p * 64 + ci0 + 16 * lh + 8);
      v16h a = __builtin_shufflevector(alo, ahi, 0, 1, 2, 3, 4, 5, 6, 7,
                                       8, 9, 10, 11, 12, 13, 14, 15);
      // B fragment: lane holds column N, K = 16*lh .. 16*lh+15 (contig channels)
      const int cb = ci0 + 16 * lh;
      v8h b0l = *reinterpret_cast<const v8h*>(brow0 + cb);
      v8h b0h = *reinterpret_cast<const v8h*>(brow0 + cb + 8);
      v16h b0 = __builtin_shufflevector(b0l, b0h, 0, 1, 2, 3, 4, 5, 6, 7,
                                        8, 9, 10, 11, 12, 13, 14, 15);
      acc0 = __builtin_amdgcn_wmma_f32_16x16x32_f16(
          false, a, false, b0, (short)0, acc0, false, false);
      v8h b1l = *reinterpret_cast<const v8h*>(brow1 + cb);
      v8h b1h = *reinterpret_cast<const v8h*>(brow1 + cb + 8);
      v16h b1 = __builtin_shufflevector(b1l, b1h, 0, 1, 2, 3, 4, 5, 6, 7,
                                        8, 9, 10, 11, 12, 13, 14, 15);
      acc1 = __builtin_amdgcn_wmma_f32_16x16x32_f16(
          false, a, false, b1, (short)0, acc1, false, false);
    }
  }

  // D layout: VGPR r, lanes0-15 -> M=r, lanes16-31 -> M=r+8; N = lane%16
  // Rows are contiguous in NHWC -> emit as b128 stores.
  float* ybase = y + ((tb * H_ + h) * W_ + w0) * 64;
  const int n0 = nb0 + lm;
  const int n1 = nb0 + 16 + lm;
  const int m0 = mBase + 8 * lh;
  v4f a0lo = __builtin_shufflevector(acc0, acc0, 0, 1, 2, 3);
  v4f a0hi = __builtin_shufflevector(acc0, acc0, 4, 5, 6, 7);
  v4f a1lo = __builtin_shufflevector(acc1, acc1, 0, 1, 2, 3);
  v4f a1hi = __builtin_shufflevector(acc1, acc1, 4, 5, 6, 7);
  *reinterpret_cast<v4f*>(ybase + n0 * 64 + m0)     = a0lo;
  *reinterpret_cast<v4f*>(ybase + n0 * 64 + m0 + 4) = a0hi;
  *reinterpret_cast<v4f*>(ybase + n1 * 64 + m0)     = a1lo;
  *reinterpret_cast<v4f*>(ybase + n1 * 64 + m0 + 4) = a1hi;
}

// ---------------------------------------------------------------------------
// Host orchestration
// ---------------------------------------------------------------------------
extern "C" void kernel_launch(void* const* d_in, const int* in_sizes, int n_in,
                              void* d_out, int out_size, void* d_ws, size_t ws_size,
                              hipStream_t stream) {
  (void)in_sizes; (void)n_in; (void)out_size; (void)ws_size;
  const float* x      = (const float*)d_in[0];
  const float* conv_w = (const float*)d_in[1];
  const float* gamma  = (const float*)d_in[2];
  const float* beta   = (const float*)d_in[3];
  const float* scale  = (const float*)d_in[4];
  float* outp         = (float*)d_out;

  char* p = (char*)d_ws;
  auto carve = [&](size_t bytes) {
    void* r = (void*)p;
    p += (bytes + 255) & ~(size_t)255;
    return r;
  };
  h16*   gw   = (h16*)  carve((size_t)8 * 64 * K_ * sizeof(h16));
  h16*   spk  = (h16*)  carve((size_t)NEL * sizeof(h16));
  float* ybuf = (float*)carve((size_t)NEL * sizeof(float));
  float* outb = (float*)carve((size_t)NEL * sizeof(float));
  float* vseq = (float*)carve((size_t)NEL * sizeof(float));
  float* part = (float*)carve((size_t)256 * 128 * sizeof(float));
  float* cA   = (float*)carve((size_t)256 * sizeof(float));
  float* cB   = (float*)carve((size_t)256 * sizeof(float));

  const int ELEM_BLOCKS = (B_ * HW_ * C_) / 256;  // 16384
  const int CONV_BLOCKS = TB_ * H_ * (W_ / 64);   // 4096

  k_prep_w<<<(8 * 64 * K_ + 255) / 256, 256, 0, stream>>>(conv_w, gw);
  k_lif_x<<<ELEM_BLOCKS, 256, 0, stream>>>(x, spk);

  auto conv = [&](int j) {
    k_conv3x3_wmma<<<CONV_BLOCKS, 256, 0, stream>>>(spk, gw + (size_t)j * 64 * K_, ybuf);
    k_stats_part<<<256, 256, 0, stream>>>(ybuf, part);
    k_stats_final<<<1, 64, 0, stream>>>(part, gamma + j * 64, beta + j * 64,
                                        scale + j * 4, cA, cB);
  };

  // sub-block 1: lif1 -> conv0/bn0 -> lif2 -> conv1/bn1 + x
  conv(0);
  k_bn_lif<<<ELEM_BLOCKS, 256, 0, stream>>>(ybuf, cA, cB, spk);
  conv(1);
  k_bn_add<<<ELEM_BLOCKS, 256, 0, stream>>>(ybuf, cA, cB, x, outb,
                                            /*resid_nchw=*/1, /*out_nchw=*/0);

  // sub-blocks 2..4: mid LIF gives spikes + v_seq residual
  for (int i = 1; i < 4; ++i) {
    k_lif_mid<<<ELEM_BLOCKS, 256, 0, stream>>>(outb, spk, vseq);
    conv(2 * i);
    k_bn_lif<<<ELEM_BLOCKS, 256, 0, stream>>>(ybuf, cA, cB, spk);
    conv(2 * i + 1);
    if (i < 3)
      k_bn_add<<<ELEM_BLOCKS, 256, 0, stream>>>(ybuf, cA, cB, vseq, outb, 0, 0);
    else
      k_bn_add<<<ELEM_BLOCKS, 256, 0, stream>>>(ybuf, cA, cB, vseq, outp, 0,
                                                /*out_nchw=*/1);
  }
}
